// SparseLayerDense_10359461118625
// MI455X (gfx1250) — matrix operand order repaired and checked by reading, not compiled, not run on hardware
//
#include <hip/hip_runtime.h>
#include <hip/hip_bf16.h>

// Problem constants (match reference)
#define IN_F   4096
#define UNITS  4096
#define BATCH  8192

typedef __bf16 bf16_t;
typedef bf16_t v8bf  __attribute__((ext_vector_type(8)));
typedef bf16_t v16bf __attribute__((ext_vector_type(16)));
typedef float  v4f   __attribute__((ext_vector_type(4)));
typedef float  v8f   __attribute__((ext_vector_type(8)));

// Workspace layout: W^T as bf16, row-major [n][k], at d_ws offset 0
#define WT_ELEMS ((size_t)UNITS * (size_t)IN_F)
#define WT_BYTES (WT_ELEMS * 2)                   // 33,554,432

// ---------------------------------------------------------------------------
// Kernel 1: zero the bf16 W^T buffer (16B vector stores)
// ---------------------------------------------------------------------------
__global__ void zero_wt_kernel(uint4* __restrict__ wt16, size_t n16) {
    size_t i = (size_t)blockIdx.x * blockDim.x + threadIdx.x;
    if (i < n16) {
        uint4 z; z.x = 0u; z.y = 0u; z.z = 0u; z.w = 0u;
        wt16[i] = z;
    }
}

// ---------------------------------------------------------------------------
// Kernel 2: scatter COO values into W^T[n][k] (bf16)
// ---------------------------------------------------------------------------
__global__ void scatter_w_kernel(const float* __restrict__ kern,
                                 const int* __restrict__ rows,
                                 const int* __restrict__ cols,
                                 bf16_t* __restrict__ wt, int nnz) {
    int i = blockIdx.x * blockDim.x + threadIdx.x;
    if (i < nnz) {
        int r = rows[i];   // k index
        int c = cols[i];   // n index
        wt[(size_t)c * IN_F + r] = (bf16_t)kern[i];
    }
}

// ---------------------------------------------------------------------------
// Fragment loaders (per-lane layouts used here; contiguous k order, same for A and B):
// A (16x32 bf16): lane run  [k+hi*16, +16),                    row M = lo
//   loaded as fp32 and packed via v_cvt_pk_bf16_f32 (co-executes with XDL)
// B (32x16 bf16): lane run  [k+hi*16, +16),                    col N = lo
// ---------------------------------------------------------------------------
__device__ __forceinline__ v16bf load_a_frag(const float* __restrict__ pa) {
    v4f a0 = *(const v4f*)(pa);
    v4f a1 = *(const v4f*)(pa + 4);
    v4f a2 = *(const v4f*)(pa + 8);
    v4f a3 = *(const v4f*)(pa + 12);
    v16bf av;
#pragma unroll
    for (int e = 0; e < 4; ++e) {
        av[e]      = (bf16_t)a0[e];
        av[e + 4]  = (bf16_t)a1[e];
        av[e + 8]  = (bf16_t)a2[e];
        av[e + 12] = (bf16_t)a3[e];
    }
    return av;
}

__device__ __forceinline__ v16bf load_b_frag(const bf16_t* __restrict__ pb) {
    v8bf b0 = *(const v8bf*)(pb);
    v8bf b1 = *(const v8bf*)(pb + 8);
    return __builtin_shufflevector(b0, b1, 0, 1, 2, 3, 4, 5, 6, 7,
                                           8, 9, 10, 11, 12, 13, 14, 15);
}

__device__ __forceinline__ void mma_step(v8f acc[4][4], const v16bf A[4], const v16bf B[4]) {
#pragma unroll
    for (int i = 0; i < 4; ++i) {
#pragma unroll
        for (int j = 0; j < 4; ++j) {
            acc[i][j] = __builtin_amdgcn_wmma_f32_16x16x32_bf16(
                /*neg_a=*/false, A[i],
                /*neg_b=*/false, B[j],
                /*c_mod=*/(short)0, acc[i][j],
                /*reuse_a=*/false, /*reuse_b=*/false);
        }
    }
}

// ---------------------------------------------------------------------------
// Kernel 3: bf16 WMMA GEMM, out[m][n] = relu( sum_k x[m][k]*wt[n][k] + bias[n] )
//   Block: 256 threads = 8 waves, 2(M) x 4(N); block tile 128 x 256.
//   Wave tile: 64 x 64 = 4x4 v_wmma_f32_16x16x32_bf16, fp32 accumulate.
//   K-loop manually unrolled x2 with double-buffered fragment registers so
//   next-step loads/cvts never overwrite VGPRs still owned by in-flight WMMAs
//   (avoids the 4-coexec WMMA->VALU WAR hazard NOPs).
// ---------------------------------------------------------------------------
__global__ __launch_bounds__(256)
void gemm_bf16_wmma_kernel(const float* __restrict__ x,
                           const bf16_t* __restrict__ wt,
                           const float* __restrict__ bias,
                           float* __restrict__ out) {
    const int lane = threadIdx.x & 31;
    const int wave = threadIdx.x >> 5;
    const int lo   = lane & 15;
    const int hi   = lane >> 4;

    const int mBase = blockIdx.y * 128 + (wave >> 2) * 64;
    const int nBase = blockIdx.x * 256 + (wave & 3) * 64;

    const float*  paBase[4];
    const bf16_t* pbBase[4];
#pragma unroll
    for (int i = 0; i < 4; ++i) {
        paBase[i] = x  + (size_t)(mBase + i * 16 + lo) * IN_F + hi * 16;
        pbBase[i] = wt + (size_t)(nBase + i * 16 + lo) * IN_F + hi * 16;
    }

    v8f acc[4][4] = {};

    v16bf A0[4], B0[4], A1[4], B1[4];
#pragma unroll
    for (int i = 0; i < 4; ++i) {
        A0[i] = load_a_frag(paBase[i]);
        B0[i] = load_b_frag(pbBase[i]);
    }

    int k = 0;
    for (; k + 64 < IN_F; k += 64) {
#pragma unroll
        for (int i = 0; i < 4; ++i) {
            A1[i] = load_a_frag(paBase[i] + k + 32);
            B1[i] = load_b_frag(pbBase[i] + k + 32);
        }
        mma_step(acc, A0, B0);
#pragma unroll
        for (int i = 0; i < 4; ++i) {
            A0[i] = load_a_frag(paBase[i] + k + 64);
            B0[i] = load_b_frag(pbBase[i] + k + 64);
        }
        mma_step(acc, A1, B1);
    }
    // Tail: k == IN_F - 64; A0/B0 hold [k, k+32)
#pragma unroll
    for (int i = 0; i < 4; ++i) {
        A1[i] = load_a_frag(paBase[i] + IN_F - 32);
        B1[i] = load_b_frag(pbBase[i] + IN_F - 32);
    }
    mma_step(acc, A0, B0);
    mma_step(acc, A1, B1);

    // Epilogue: bias + relu, streamed (non-temporal) fp32 stores so the
    // write-once 134MB output does not evict x/Wt from L2.
#pragma unroll
    for (int j = 0; j < 4; ++j) {
        const int n = nBase + j * 16 + lo;
        const float bv = bias[n];
#pragma unroll
        for (int i = 0; i < 4; ++i) {
#pragma unroll
            for (int v = 0; v < 8; ++v) {
                const int m = mBase + i * 16 + hi * 8 + v;
                float val = acc[i][j][v] + bv;
                val = val > 0.0f ? val : 0.0f;
                __builtin_nontemporal_store(val, &out[(size_t)m * UNITS + n]);
            }
        }
    }
}

// ---------------------------------------------------------------------------
// Host launcher
// ---------------------------------------------------------------------------
extern "C" void kernel_launch(void* const* d_in, const int* in_sizes, int n_in,
                              void* d_out, int out_size, void* d_ws, size_t ws_size,
                              hipStream_t stream) {
    const float* x    = (const float*)d_in[0];   // [BATCH, IN_F] fp32
    const float* kern = (const float*)d_in[1];   // [NNZ] fp32
    const float* bias = (const float*)d_in[2];   // [UNITS] fp32
    const int*   rows = (const int*)d_in[3];     // [NNZ] int32 (k index)
    const int*   cols = (const int*)d_in[4];     // [NNZ] int32 (n index)
    float*       out  = (float*)d_out;           // [BATCH, UNITS] fp32

    const int nnz = in_sizes[1];

    bf16_t* wt = (bf16_t*)d_ws;                  // 33.5 MB

    // 1) Zero W^T (bf16) — must run every call (non-scattered entries are 0)
    {
        size_t n16 = WT_BYTES / 16;
        zero_wt_kernel<<<(unsigned)((n16 + 255) / 256), 256, 0, stream>>>((uint4*)wt, n16);
    }
    // 2) Scatter sparse weights into W^T
    scatter_w_kernel<<<(nnz + 255) / 256, 256, 0, stream>>>(kern, rows, cols, wt, nnz);

    // 3) Dense bf16 WMMA GEMM with fused x-conversion, bias + relu
    dim3 grid(UNITS / 256, BATCH / 128);
    gemm_bf16_wmma_kernel<<<grid, 256, 0, stream>>>(x, wt, bias, out);
}
